// CasulaSelfAttention_57655640981927
// MI455X (gfx1250) — compile-verified, matrix operand order checked by reading, not run on hardware
//
#include <hip/hip_runtime.h>

// ---------------------------------------------------------------------------
// Causal self-attention for MI455X (gfx1250, wave32, WMMA bf16).
// B=4, T=2048, C=1024, H=16, D=64.
// ---------------------------------------------------------------------------

typedef __attribute__((ext_vector_type(16))) __bf16       v16bf;
typedef __attribute__((ext_vector_type(8)))  float        v8f;
typedef __attribute__((ext_vector_type(8)))  unsigned int v8u;

static constexpr int Bn = 4;
static constexpr int Tn = 2048;
static constexpr int Cn = 1024;
static constexpr int Hn = 16;
static constexpr int Dn = 64;

// ---------------- helpers ----------------

static __device__ __forceinline__ unsigned short f2bf(float f) {
  unsigned int x = __float_as_uint(f);
  x += 0x7fffu + ((x >> 16) & 1u);          // round-to-nearest-even
  return (unsigned short)(x >> 16);
}

// Load a 16x32 bf16 A/B fragment from row-major memory.
// Lane L holds row (L&15); K pairs: lanes 0-15 -> {0..7,16..23},
// lanes 16-31 -> {8..15,24..31}; two bf16 per dword along K.
static __device__ __forceinline__ v16bf load_frag_g(const unsigned short* base, int ld) {
  const int lane = threadIdx.x & 31;
  const unsigned short* p = base + (size_t)(lane & 15) * ld + ((lane >> 4) << 3);
  v8u u;
#pragma unroll
  for (int j = 0; j < 4; ++j)
    u[j] = *reinterpret_cast<const unsigned int*>(p + 2 * j);
#pragma unroll
  for (int j = 0; j < 4; ++j)
    u[4 + j] = *reinterpret_cast<const unsigned int*>(p + 16 + 2 * j);
  return __builtin_bit_cast(v16bf, u);
}

// Same fragment pattern out of a row-major bf16 LDS tile with leading dim ld.
static __device__ __forceinline__ v16bf load_frag_lds(const unsigned short* base, int ld) {
  const int lane = threadIdx.x & 31;
  const unsigned short* p = base + (lane & 15) * ld + ((lane >> 4) << 3);
  v8u u;
#pragma unroll
  for (int j = 0; j < 4; ++j) {
    unsigned int lo = p[2 * j], hi = p[2 * j + 1];
    u[j] = lo | (hi << 16);
  }
#pragma unroll
  for (int j = 0; j < 4; ++j) {
    unsigned int lo = p[16 + 2 * j], hi = p[16 + 2 * j + 1];
    u[4 + j] = lo | (hi << 16);
  }
  return __builtin_bit_cast(v16bf, u);
}

static __device__ __forceinline__ v8f wmma_bf16(v16bf a, v16bf b, v8f c) {
  return __builtin_amdgcn_wmma_f32_16x16x32_bf16(false, a, false, b, (short)0, c,
                                                 false, false);
}

// ---- DPP cross-lane max within 16-lane halves (no LDS traffic) ----

static constexpr int dpp8_sel(int a, int b, int c, int d, int e, int f, int g, int h) {
  return a | (b << 3) | (c << 6) | (d << 9) | (e << 12) | (f << 15) | (g << 18) |
         (h << 21);
}
static constexpr int SEL_X1 = dpp8_sel(1, 0, 3, 2, 5, 4, 7, 6);
static constexpr int SEL_X2 = dpp8_sel(2, 3, 0, 1, 6, 7, 4, 5);
static constexpr int SEL_X4 = dpp8_sel(4, 5, 6, 7, 0, 1, 2, 3);
static constexpr int ROW_ROR8 = 0x128;   // DPP16 ROW_ROR:8 == xor8 within 16 lanes

template <int SEL>
static __device__ __forceinline__ float dpp8f(float v) {
  return __int_as_float(__builtin_amdgcn_mov_dpp8(__float_as_int(v), SEL));
}
static __device__ __forceinline__ float rowmax16(float t) {
  t = fmaxf(t, dpp8f<SEL_X1>(t));
  t = fmaxf(t, dpp8f<SEL_X2>(t));
  t = fmaxf(t, dpp8f<SEL_X4>(t));
  t = fmaxf(t, __int_as_float(__builtin_amdgcn_mov_dpp(__float_as_int(t), ROW_ROR8,
                                                       0xf, 0xf, true)));
  return t;
}

// ---------------- conversion / transpose ----------------

__global__ void k_f32_to_bf16(const float* __restrict__ in,
                              unsigned short* __restrict__ out, int n) {
  int i = blockIdx.x * blockDim.x + threadIdx.x;
  if (i < n) out[i] = f2bf(in[i]);
}

// in: [K][N] f32 row-major  ->  out: [N][K] bf16 row-major (i.e. B^T)
__global__ void k_transpose_bf16(const float* __restrict__ in,
                                 unsigned short* __restrict__ out, int K, int N) {
  int i = blockIdx.x * blockDim.x + threadIdx.x;
  if (i < K * N) {
    int n = i / K, k = i - n * K;
    out[i] = f2bf(in[(size_t)k * N + n]);
  }
}

// ---------------- WMMA GEMM: Out = A[M,K] * Bt[N,K]^T + bias ----------------
// EPI 0: QKV epilogue (scatter bf16 q/k and transposed v, pre-scale q)
// EPI 1: f32 output (final projection)

template <int EPI>
__global__ __launch_bounds__(256) void k_gemm(
    const unsigned short* __restrict__ A, const unsigned short* __restrict__ Bt,
    const float* __restrict__ bias, int K, float* __restrict__ outf,
    unsigned short* __restrict__ qbf, unsigned short* __restrict__ kbf,
    unsigned short* __restrict__ vtbf, int Ntot) {
  const int lane = threadIdx.x & 31;
  const int w = threadIdx.x >> 5;
  const int row0 = blockIdx.x * 64 + (w >> 2) * 32;
  const int col0 = blockIdx.y * 256 + (w & 3) * 64;

  v8f acc[2][4] = {};

  for (int k0 = 0; k0 < K; k0 += 32) {
    v16bf a0 = load_frag_g(A + (size_t)row0 * K + k0, K);
    v16bf a1 = load_frag_g(A + (size_t)(row0 + 16) * K + k0, K);
#pragma unroll
    for (int j = 0; j < 4; ++j) {
      v16bf b = load_frag_g(Bt + (size_t)(col0 + j * 16) * K + k0, K);
      acc[0][j] = wmma_bf16(a0, b, acc[0][j]);
      acc[1][j] = wmma_bf16(a1, b, acc[1][j]);
    }
  }

  const int half = lane >> 4, nl = lane & 15;
#pragma unroll
  for (int j = 0; j < 4; ++j) {
    const int col = col0 + j * 16 + nl;
    const float bv = bias[col];
    if (EPI == 0) {
      const int which = col >> 10;          // 0:q 1:k 2:v
      const int c = col & (Cn - 1);
      const int h = c >> 6, d = c & (Dn - 1);
#pragma unroll
      for (int i = 0; i < 2; ++i) {
#pragma unroll
        for (int r = 0; r < 8; ++r) {
          const int row = row0 + i * 16 + r + 8 * half;    // global token index
          const int b = row >> 11;                          // /T
          const int t = row & (Tn - 1);
          float v = acc[i][j][r] + bv;
          const size_t bh = (size_t)(b * Hn + h);
          if (which == 0) {
            v *= 0.125f;                                    // 1/sqrt(D)
            qbf[(bh * Tn + t) * Dn + d] = f2bf(v);
          } else if (which == 1) {
            kbf[(bh * Tn + t) * Dn + d] = f2bf(v);
          } else {
            vtbf[(bh * Dn + d) * Tn + t] = f2bf(v);         // V transposed
          }
        }
      }
    } else {
#pragma unroll
      for (int i = 0; i < 2; ++i) {
#pragma unroll
        for (int r = 0; r < 8; ++r) {
          const int row = row0 + i * 16 + r + 8 * half;
          outf[(size_t)row * Ntot + col] = acc[i][j][r] + bv;
        }
      }
    }
  }
}

// ---------------- flash attention (online softmax, 64-key blocks) ----------
// grid.x = B*H (64), grid.y = T/128 (16); 8 waves/block, one 16-row Q tile per
// wave.  Row sums accumulate through an all-ones WMMA; row max uses DPP.

template <bool MASK>
static __device__ __forceinline__ void attn_block(
    int k0, int q0, const unsigned short* kbase0, const unsigned short* vbase,
    unsigned short* sp, const v16bf& aq0, const v16bf& aq1, const v16bf& ones,
    v8f o[4], v8f& osum, float mrun[8]) {
  const int lane = threadIdx.x & 31;
  const int half = lane >> 4, nl = lane & 15;

  // S = Q * K^T : four 16x16 tiles covering 64 keys
  v8f s[4];
#pragma unroll
  for (int c = 0; c < 4; ++c) {
    const unsigned short* kb = kbase0 + (size_t)(k0 + c * 16) * Dn;
    v8f sc = {};
    sc = wmma_bf16(aq0, load_frag_g(kb, Dn), sc);
    sc = wmma_bf16(aq1, load_frag_g(kb + 32, Dn), sc);
    s[c] = sc;
  }

  float al[8], p[4][8];
#pragma unroll
  for (int r = 0; r < 8; ++r) {
    const int qrow = q0 + r + 8 * half;
    float v[4];
#pragma unroll
    for (int c = 0; c < 4; ++c) {
      v[c] = s[c][r];
      if (MASK) v[c] = (k0 + c * 16 + nl > qrow) ? -__builtin_inff() : v[c];
    }
    float t = fmaxf(fmaxf(v[0], v[1]), fmaxf(v[2], v[3]));
    t = rowmax16(t);
    const float mn = fmaxf(mrun[r], t);
    al[r] = __expf(mrun[r] - mn);
    mrun[r] = mn;
#pragma unroll
    for (int c = 0; c < 4; ++c) p[c][r] = __expf(v[c] - mn);
  }

  // rescale running state
#pragma unroll
  for (int j = 0; j < 4; ++j)
#pragma unroll
    for (int r = 0; r < 8; ++r) o[j][r] *= al[r];
#pragma unroll
  for (int r = 0; r < 8; ++r) osum[r] *= al[r];

  // stage P (C-layout -> LDS [16][64]) then reload as two A fragments
#pragma unroll
  for (int r = 0; r < 8; ++r) {
    const int m = r + 8 * half;
#pragma unroll
    for (int c = 0; c < 4; ++c) sp[m * 64 + c * 16 + nl] = f2bf(p[c][r]);
  }
  asm volatile("s_wait_dscnt 0x0" ::: "memory");
  const v16bf ap0 = load_frag_lds(sp, 64);        // keys k0   .. k0+31
  const v16bf ap1 = load_frag_lds(sp + 32, 64);   // keys k0+32.. k0+63

  // O += P * V  (V^T row-major: rows = d, cols = t);  osum += P * 1
#pragma unroll
  for (int j = 0; j < 4; ++j) {
    const unsigned short* vb = vbase + (size_t)(j * 16) * Tn + k0;
    o[j] = wmma_bf16(ap0, load_frag_g(vb, Tn), o[j]);
    o[j] = wmma_bf16(ap1, load_frag_g(vb + 32, Tn), o[j]);
  }
  osum = wmma_bf16(ap0, ones, osum);
  osum = wmma_bf16(ap1, ones, osum);
}

__global__ __launch_bounds__(256) void k_attn(const unsigned short* __restrict__ qbf,
                                              const unsigned short* __restrict__ kbf,
                                              const unsigned short* __restrict__ vtbf,
                                              unsigned short* __restrict__ ybf) {
  __shared__ unsigned short sP[8][16 * 64];   // per-wave P staging, 2 KB each

  const int lane = threadIdx.x & 31;
  const int w = threadIdx.x >> 5;
  const int bh = blockIdx.x;
  const int q0 = blockIdx.y * 128 + w * 16;
  const int half = lane >> 4, nl = lane & 15;

  const unsigned short* qbase = qbf + ((size_t)bh * Tn + q0) * Dn;
  const v16bf aq0 = load_frag_g(qbase, Dn);        // d 0..31
  const v16bf aq1 = load_frag_g(qbase + 32, Dn);   // d 32..63

  v8u ou;
#pragma unroll
  for (int j = 0; j < 8; ++j) ou[j] = 0x3f803f80u;  // bf16 1.0 pairs
  const v16bf ones = __builtin_bit_cast(v16bf, ou);

  v8f o[4] = {};
  v8f osum = {};
  float mrun[8];
#pragma unroll
  for (int r = 0; r < 8; ++r) mrun[r] = -__builtin_inff();

  const unsigned short* kbase0 = kbf + (size_t)bh * Tn * Dn;
  const unsigned short* vbase = vtbf + (size_t)bh * Dn * Tn;
  unsigned short* sp = &sP[w][0];

  // unmasked full 64-key blocks, then exactly one masked diagonal block
  const int nfull = (q0 >= 64) ? (((q0 - 63) >> 6) + 1) : 0;
  for (int i = 0; i < nfull; ++i)
    attn_block<false>(i * 64, q0, kbase0, vbase, sp, aq0, aq1, ones, o, osum, mrun);
  attn_block<true>(nfull * 64, q0, kbase0, vbase, sp, aq0, aq1, ones, o, osum, mrun);

  // normalize and write y (bf16, [B,T,C] with c = h*D + d).  Every column of
  // osum equals the row sum, so each lane reads its own slot directly.
  const int b = bh >> 4, h = bh & 15;
#pragma unroll
  for (int j = 0; j < 4; ++j) {
#pragma unroll
    for (int r = 0; r < 8; ++r) {
      const int trow = q0 + r + 8 * half;
      const int d = j * 16 + nl;
      const float v = o[j][r] / osum[r];
      ybf[((size_t)b * Tn + trow) * Cn + h * Dn + d] = f2bf(v);
    }
  }
}

// ---------------- launch ----------------

extern "C" void kernel_launch(void* const* d_in, const int* in_sizes, int n_in,
                              void* d_out, int out_size, void* d_ws, size_t ws_size,
                              hipStream_t stream) {
  const float* x      = (const float*)d_in[0];   // [B,T,C]
  const float* w_attn = (const float*)d_in[1];   // [C,3C]
  const float* b_attn = (const float*)d_in[2];   // [3C]
  const float* w_proj = (const float*)d_in[3];   // [C,C]
  const float* b_proj = (const float*)d_in[4];   // [C]
  float* out = (float*)d_out;                    // [B,T,C]

  const size_t SZ_XBF = (size_t)Bn * Tn * Cn * 2;       // 16 MB
  const size_t SZ_WA  = (size_t)Cn * 3 * Cn * 2;        //  6 MB
  const size_t SZ_WP  = (size_t)Cn * Cn * 2;            //  2 MB
  const size_t SZ_QKV = (size_t)Bn * Hn * Tn * Dn * 2;  // 16 MB each

  char* p = (char*)d_ws;
  unsigned short* x_bf = (unsigned short*)p; p += SZ_XBF;
  unsigned short* waT  = (unsigned short*)p; p += SZ_WA;
  unsigned short* wpT  = (unsigned short*)p; p += SZ_WP;
  unsigned short* qb   = (unsigned short*)p; p += SZ_QKV;
  unsigned short* kb   = (unsigned short*)p; p += SZ_QKV;
  unsigned short* vt   = (unsigned short*)p; p += SZ_QKV;
  unsigned short* yb   = x_bf;   // reuse: x_bf dead after QKV gemm

  const int nx = Bn * Tn * Cn;
  k_f32_to_bf16<<<nx / 256, 256, 0, stream>>>(x, x_bf, nx);
  k_transpose_bf16<<<(Cn * 3 * Cn) / 256, 256, 0, stream>>>(w_attn, waT, Cn, 3 * Cn);
  k_transpose_bf16<<<(Cn * Cn) / 256, 256, 0, stream>>>(w_proj, wpT, Cn, Cn);

  // QKV: [8192,1024] x [1024,3072] -> q/k/v bf16
  k_gemm<0><<<dim3((Bn * Tn) / 64, (3 * Cn) / 256), 256, 0, stream>>>(
      x_bf, waT, b_attn, Cn, nullptr, qb, kb, vt, 3 * Cn);

  // attention
  k_attn<<<dim3(Bn * Hn, Tn / 128), 256, 0, stream>>>(qb, kb, vt, yb);

  // projection: [8192,1024] x [1024,1024] + bias -> f32 out
  k_gemm<1><<<dim3((Bn * Tn) / 64, Cn / 256), 256, 0, stream>>>(
      yb, wpT, b_proj, Cn, out, nullptr, nullptr, nullptr, Cn);
}
